// BiLSTM_46557445489256
// MI455X (gfx1250) — matrix operand order repaired and checked by reading, not compiled, not run on hardware
//
#include <hip/hip_runtime.h>
#include <hip/hip_fp16.h>
#include <math.h>

// ---------------------------------------------------------------------------
// BiLSTM for MI455X (gfx1250, wave32, WMMA + TDM).
//   x:(16,2048,256)f32, W:(256,1024), R:(256,1024), bc:(256) per direction.
//   out:(16,2048,512)f32  [forward | backward-in-iteration-order]
// Strategy: fused persistent recurrent kernel, one workgroup per direction.
//   - f16 WMMA 16x16x32, fp32 accumulate.
//   - h kept in LDS in A-fragment layout; c in registers.
//   - x pre-packed into A-fragments; per-step 8KB block DMA'd into an LDS
//     double buffer by the Tensor Data Mover (tensor_load_to_lds +
//     s_wait_tensorcnt), two steps ahead of consumption.
//   - W/R pre-packed into B-fragments, streamed from L2 every step
//     (N-tile ownership rotates with t to defeat invariant-load hoisting).
//   - x@W for step t+1 software-pipelined behind the gate phase of step t.
// ---------------------------------------------------------------------------

typedef __attribute__((ext_vector_type(16))) _Float16     v16h;
typedef __attribute__((ext_vector_type(8)))  float        v8f;
typedef __attribute__((ext_vector_type(4)))  unsigned int v4u;
typedef __attribute__((ext_vector_type(8)))  int          v8i;
typedef __attribute__((ext_vector_type(4)))  int          v4i;

#define TT 2048
#define BB 16
#define DD 256
#define UU 256
#define GG 1024   // 4*U
#define NTILES 64 // GG/16
#define KTILES 8  // 256/32

#define XFRAG_ELEMS ((size_t)TT * KTILES * 32 * 16)     // 8,388,608 f16 = 16 MiB
#define BFRAG_ELEMS ((size_t)KTILES * NTILES * 32 * 16) // 262,144 f16 = 512 KiB
#define XBLK_HALF   (KTILES * 32 * 16)                  // 4096 f16 = 8 KiB / step

// Documented 16-bit A-matrix 16x32 fragment layout (lane, elem) -> k offset.
__device__ __host__ __forceinline__ int a_klocal(int lane, int e) {
  int v = e >> 1, p = e & 1;
  return ((v >= 4) ? 16 : 0) + ((lane >= 16) ? 8 : 0) + ((v & 3) * 2) + p;
}

// ---------------- pack kernels ----------------

// x[b=m][t][32*kt + k] -> xf[((t*8+kt)*32+lane)*16 + e]  (A-fragment order)
__global__ void pack_x_frags(const float* __restrict__ x,
                             _Float16* __restrict__ xf) {
  int g = blockIdx.x * blockDim.x + threadIdx.x;  // (t*KTILES + kt)*32 + lane
  if (g >= TT * KTILES * 32) return;
  int lane = g & 31;
  int kt   = (g >> 5) & 7;
  int t    = g >> 8;
  int m    = lane & 15;  // batch row
  _Float16* dst = xf + (size_t)g * 16;
  const float* src = x + ((size_t)m * TT + t) * DD + kt * 32;
#pragma unroll
  for (int e = 0; e < 16; ++e) dst[e] = (_Float16)src[a_klocal(lane, e)];
}

// B-fragment (32x16, K x N): lanes 0-15 hold K=0..7,16..23 (col=lane),
// lanes 16-31 hold K=8..15,24..31 (col=lane-16); elem e <-> local k.
__global__ void pack_b_frags(const float* __restrict__ src,
                             _Float16* __restrict__ dst) {
  int g = blockIdx.x * blockDim.x + threadIdx.x;  // (kt*NTILES + nt)*32 + lane
  if (g >= KTILES * NTILES * 32) return;
  int lane = g & 31;
  int nt   = (g >> 5) & 63;
  int kt   = g >> 11;
  int n     = lane & 15;
  int kbase = (lane >= 16) ? 8 : 0;
  _Float16* d = dst + (size_t)g * 16;
  const float* s = src + (size_t)(kt * 32 + kbase) * GG + nt * 16 + n;
#pragma unroll
  for (int e = 0; e < 16; ++e) d[e] = (_Float16)s[(size_t)(e + ((e >= 8) ? 8 : 0)) * GG];
}

// ---------------- TDM helper: DMA one 8 KiB fragment block to LDS ----------

__device__ __forceinline__ void tdm_load_xblk(unsigned lds_off,
                                              unsigned long long gaddr) {
  // D# group 0 (128b): count=1 | lds_addr | global_addr[56:0] | type=2
  v4u g0;
  g0[0] = 1u;
  g0[1] = lds_off;
  g0[2] = (unsigned)(gaddr & 0xFFFFFFFFull);
  g0[3] = ((unsigned)(gaddr >> 32) & 0x01FFFFFFu) | 0x80000000u;
  // D# group 1 (256b): no multicast, data_size=3 (8B units),
  //   tensor_dim0 = tile_dim0 = 1024 (=> 8 KiB), tensor_dim1 = tile_dim1 = 1.
  v8i g1;
  g1[0] = 0x00030000;                   // workgroup_mask=0, data_size=3
  g1[1] = (int)0x04000000u;             // tensor_dim0[15:0]=1024 in [31:16]
  g1[2] = (int)0x00010000u;             // tensor_dim0 hi=0 | tensor_dim1 lo=1
  g1[3] = (int)0x04000000u;             // tensor_dim1 hi=0 | tile_dim0=1024
  g1[4] = 1;                            // tile_dim1=1, tile_dim2=0
  g1[5] = 1024;                         // tensor_dim0_stride[31:0]
  g1[6] = (int)0x04000000u;             // stride0 hi=0 | stride1 lo=1024
  g1[7] = 0;
  v4i gz = {0, 0, 0, 0};                // <=2-D: groups 2/3 disabled (zero)
#if __clang_major__ >= 23
  v8i gz8 = {0, 0, 0, 0, 0, 0, 0, 0};
  __builtin_amdgcn_tensor_load_to_lds(g0, g1, gz, gz, gz8, 0);
#else
  __builtin_amdgcn_tensor_load_to_lds(g0, g1, gz, gz, 0);
#endif
}

// ---------------- persistent recurrent kernel ----------------

__device__ __forceinline__ float sigmoidf_(float x) {
  return 1.0f / (1.0f + __expf(-x));
}

__launch_bounds__(1024)
__global__ void bilstm_recurrent(const _Float16* __restrict__ xfrag,
                                 const _Float16* __restrict__ wfrag_f,
                                 const _Float16* __restrict__ rfrag_f,
                                 const float* __restrict__ bcf,
                                 const _Float16* __restrict__ wfrag_b,
                                 const _Float16* __restrict__ rfrag_b,
                                 const float* __restrict__ bcb,
                                 float* __restrict__ out) {
  const int dir  = blockIdx.x;  // 0 = forward, 1 = backward
  const int tid  = threadIdx.x;
  const int lane = tid & 31;
  const int wave = tid >> 5;    // 0..31

  const v16h* Wf = (const v16h*)(dir ? wfrag_b : wfrag_f);
  const v16h* Rf = (const v16h*)(dir ? rfrag_b : rfrag_f);
  const float* bc = dir ? bcb : bcf;

  // h in A-fragment layout (next-step WMMA operand), 8 KiB
  alignas(32) __shared__ _Float16 hfrag[KTILES][32][16];
  // double-buffered TDM staging for x A-fragments, 16 KiB
  alignas(32) __shared__ _Float16 xstage[2][XBLK_HALF];
  // gate pre-activations g = xW + hR, f16, padded vs bank conflicts (~32.5 KiB)
  __shared__ _Float16 gbuf[16][1040];
  __shared__ float bcs[UU];

  for (int i = tid; i < KTILES * 32 * 16; i += 1024)
    ((_Float16*)hfrag)[i] = (_Float16)0.0f;
  if (tid < UU) bcs[tid] = bc[tid];

  // scalar (per-wave uniform) predicate for the single DMA-issuing wave
  const bool dma_wave = (__builtin_amdgcn_readfirstlane(tid) >> 5) == 0;

  // time index of the packed-x block consumed at logical step tt
  auto xtime = [&](int tt) { return dir ? (TT - 1 - tt) : tt; };
  // N-tile pair owned by this wave at timestep tt (rotates to keep weight
  // loads t-dependent -> streamed from L2 instead of hoisted into registers)
  auto tile_of = [&](int tt) { return ((wave + tt) & 31) * 2; };

  // elementwise role: this thread owns unit u for batches bq*4 .. bq*4+3
  const int u  = tid & 255;
  const int bq = tid >> 8;  // 0..3
  float creg[4] = {0.0f, 0.0f, 0.0f, 0.0f};

  // Prologue DMAs: stage x blocks for steps 0 and 1.
  if (dma_wave) {
    tdm_load_xblk((unsigned)(unsigned long long)(void*)&xstage[0][0],
                  (unsigned long long)(const void*)(xfrag +
                      (size_t)xtime(0) * XBLK_HALF));
    tdm_load_xblk((unsigned)(unsigned long long)(void*)&xstage[1][0],
                  (unsigned long long)(const void*)(xfrag +
                      (size_t)xtime(1) * XBLK_HALF));
  }
  __builtin_amdgcn_s_wait_tensorcnt(0);
  __syncthreads();

  const v8f vzero = {0, 0, 0, 0, 0, 0, 0, 0};
  v8f acc0 = vzero, acc1 = vzero;

  // Prologue: acc = xW(t=0) from staged fragments, tiles used at t=0.
  {
    const int n0 = tile_of(0), n1 = n0 + 1;
#pragma unroll
    for (int kt = 0; kt < KTILES; ++kt) {
      v16h xa = *(const v16h*)&xstage[0][(kt * 32 + lane) * 16];
      v16h w0 = Wf[(kt * NTILES + n0) * 32 + lane];
      v16h w1 = Wf[(kt * NTILES + n1) * 32 + lane];
      acc0 = __builtin_amdgcn_wmma_f32_16x16x32_f16(false, xa, false, w0,
                                                    (short)0, acc0, false, false);
      acc1 = __builtin_amdgcn_wmma_f32_16x16x32_f16(false, xa, false, w1,
                                                    (short)0, acc1, false, false);
    }
  }
  __syncthreads();

  const int grow = (lane >= 16) ? 8 : 0;  // C-fragment row offset
  const int gsub = lane & 15;             // C-fragment column within tile

  for (int t = 0; t < TT; ++t) {
    const int n0 = tile_of(t), n1 = n0 + 1;  // tiles matching acc contents

    // ---- g += h @ R (the serial critical path: 8 chained WMMAs/acc) ----
#pragma unroll
    for (int kt = 0; kt < KTILES; ++kt) {
      v16h ha = *(const v16h*)(&hfrag[kt][lane][0]);
      v16h r0 = Rf[(kt * NTILES + n0) * 32 + lane];
      v16h r1 = Rf[(kt * NTILES + n1) * 32 + lane];
      acc0 = __builtin_amdgcn_wmma_f32_16x16x32_f16(false, ha, false, r0,
                                                    (short)0, acc0, false, false);
      acc1 = __builtin_amdgcn_wmma_f32_16x16x32_f16(false, ha, false, r1,
                                                    (short)0, acc1, false, false);
    }
    // scatter C tiles (documented 16x16 f32 C layout) into gbuf
#pragma unroll
    for (int r = 0; r < 8; ++r) {
      gbuf[r + grow][n0 * 16 + gsub] = (_Float16)acc0[r];
      gbuf[r + grow][n1 * 16 + gsub] = (_Float16)acc1[r];
    }
    __syncthreads();

    // ---- TDM: stage x block for step t+2 into the buffer last read at t-1
    if (dma_wave && (t + 2 < TT)) {
      tdm_load_xblk((unsigned)(unsigned long long)(void*)&xstage[t & 1][0],
                    (unsigned long long)(const void*)(xfrag +
                        (size_t)xtime(t + 2) * XBLK_HALF));
    }

    // ---- software pipeline: xW(t+1) (independent of h) ----
    acc0 = vzero; acc1 = vzero;
    if (t + 1 < TT) {
      const int m0 = tile_of(t + 1), m1 = m0 + 1;
      const _Float16* xs = &xstage[(t + 1) & 1][0];
#pragma unroll
      for (int kt = 0; kt < KTILES; ++kt) {
        v16h xa = *(const v16h*)&xs[(kt * 32 + lane) * 16];
        v16h w0 = Wf[(kt * NTILES + m0) * 32 + lane];
        v16h w1 = Wf[(kt * NTILES + m1) * 32 + lane];
        acc0 = __builtin_amdgcn_wmma_f32_16x16x32_f16(false, xa, false, w0,
                                                      (short)0, acc0, false, false);
        acc1 = __builtin_amdgcn_wmma_f32_16x16x32_f16(false, xa, false, w1,
                                                      (short)0, acc1, false, false);
      }
      // warm L0/L2 for next step's rotated R tiles (global_prefetch_b8)
      __builtin_prefetch((const void*)&Rf[(m0 * 32) + lane], 0, 1);
      __builtin_prefetch((const void*)&Rf[(7 * NTILES + m1) * 32 + lane], 0, 1);
    }

    // ---- gates + state update (reference quirks preserved) ----
#pragma unroll
    for (int j = 0; j < 4; ++j) {
      const int b = bq * 4 + j;
      float gi = (float)gbuf[b][u];
      float gf = (float)gbuf[b][256 + u];
      float go = (float)gbuf[b][512 + u];
      float gc = (float)gbuf[b][768 + u] + bcs[u];
      float i_ = sigmoidf_(gi);
      float f_ = sigmoidf_(gf);
      float o_ = sigmoidf_(go);
      float cd = tanhf(gc);
      float cn = sigmoidf_(f_ * creg[j] + i_ * cd);   // quirk: sigmoid cell
      float h  = tanhf(cn) * o_;
      creg[j] = cn;
      out[((size_t)b * TT + t) * (2 * UU) + (size_t)dir * UU + u] = h;
      // write h back into A-fragment layout for next step
      int kt_ = u >> 5;
      int k   = u & 31;
      int k2  = k & 15;
      int lw  = (b) + ((k2 >= 8) ? 16 : 0);
      int j2  = k2 - ((k2 >= 8) ? 8 : 0);
      int v   = ((k >= 16) ? 4 : 0) + (j2 >> 1);
      int e   = v * 2 + (j2 & 1);
      hfrag[kt_][lw][e] = (_Float16)h;
    }
    // drain this wave's TDM ops so barrier2 happens-after the staged data
    __builtin_amdgcn_s_wait_tensorcnt(0);
    __syncthreads();
  }
}

// ---------------- host launcher ----------------

extern "C" void kernel_launch(void* const* d_in, const int* in_sizes, int n_in,
                              void* d_out, int out_size, void* d_ws, size_t ws_size,
                              hipStream_t stream) {
  (void)in_sizes; (void)n_in; (void)out_size; (void)ws_size;
  const float* x    = (const float*)d_in[0];
  const float* W_f  = (const float*)d_in[1];
  const float* R_f  = (const float*)d_in[2];
  const float* bc_f = (const float*)d_in[3];
  const float* W_b  = (const float*)d_in[4];
  const float* R_b  = (const float*)d_in[5];
  const float* bc_b = (const float*)d_in[6];
  float* out = (float*)d_out;

  _Float16* xf  = (_Float16*)d_ws;               // 16 MiB
  _Float16* wff = xf  + XFRAG_ELEMS;             // 512 KiB each
  _Float16* rff = wff + BFRAG_ELEMS;
  _Float16* wfb = rff + BFRAG_ELEMS;
  _Float16* rfb = wfb + BFRAG_ELEMS;             // total ~18 MiB

  const int xthreads = TT * KTILES * 32;
  pack_x_frags<<<(xthreads + 255) / 256, 256, 0, stream>>>(x, xf);
  const int bthreads = KTILES * NTILES * 32;
  const int bblocks  = (bthreads + 255) / 256;
  pack_b_frags<<<bblocks, 256, 0, stream>>>(W_f, wff);
  pack_b_frags<<<bblocks, 256, 0, stream>>>(R_f, rff);
  pack_b_frags<<<bblocks, 256, 0, stream>>>(W_b, wfb);
  pack_b_frags<<<bblocks, 256, 0, stream>>>(R_b, rfb);

  bilstm_recurrent<<<2, 1024, 0, stream>>>(xf, wff, rff, bc_f,
                                           wfb, rfb, bc_b, out);
}
